// Decoder_15187004358874
// MI455X (gfx1250) — matrix operand order repaired and checked by reading, not compiled, not run on hardware
//
#include <hip/hip_runtime.h>
#include <math.h>

// ---------------- problem constants ----------------
static constexpr int kB   = 256;
static constexpr int kN   = 64;
static constexpr int kD   = 512;
static constexpr int kH   = 8;
static constexpr int kHD  = 64;
static constexpr int kNB  = 4;
static constexpr int kBN  = kB * kN;        // 16384 tokens
static constexpr int kA   = 64;

// ---------------- WMMA types ----------------
typedef __attribute__((ext_vector_type(16))) __bf16          v16bf;
typedef __attribute__((ext_vector_type(8)))  float           v8f;
typedef __attribute__((ext_vector_type(8)))  unsigned short  ushort8;

union BF16Frag {
  v16bf   v;
  unsigned short u[16];
  ushort8 h[2];
};

__device__ __forceinline__ unsigned short f2bf(float f) {
  unsigned int x = __float_as_uint(f);
  x += 0x7fffu + ((x >> 16) & 1u);          // round-to-nearest-even
  return (unsigned short)(x >> 16);
}

__device__ __forceinline__ float gelu_exact(float x) {
  return 0.5f * x * (1.0f + erff(x * 0.70710678118654752f));
}

__device__ __forceinline__ v8f vzero8() {
  v8f z;
#pragma unroll
  for (int i = 0; i < 8; ++i) z[i] = 0.0f;
  return z;
}

// 128-thread (4-wave) block reduce, wave32 shuffles + LDS combine
__device__ __forceinline__ float block128_sum(float v, float* sh) {
#pragma unroll
  for (int off = 16; off > 0; off >>= 1) v += __shfl_xor(v, off, 32);
  const int w = threadIdx.x >> 5;
  if ((threadIdx.x & 31) == 0) sh[w] = v;
  __syncthreads();
  v = sh[0] + sh[1] + sh[2] + sh[3];
  __syncthreads();
  return v;
}

// ---------------------------------------------------------------------------
// Weight prep: fp32 [K,N] row-major (torch [in,out]) -> bf16 [N,K] row-major
// so a B-matrix fragment (one output column over 16 K values) is contiguous.
// ---------------------------------------------------------------------------
__global__ void wconv_kernel(const float* __restrict__ src,
                             unsigned short* __restrict__ dst,
                             int K, int N, int nmats) {
  const size_t per   = (size_t)K * N;
  const size_t total = per * (size_t)nmats;
  for (size_t i = (size_t)blockIdx.x * blockDim.x + threadIdx.x; i < total;
       i += (size_t)gridDim.x * blockDim.x) {
    size_t m = i / per;
    size_t r = i - m * per;
    size_t n = r / (size_t)K;
    size_t k = r - n * (size_t)K;
    dst[i] = f2bf(src[m * per + k * (size_t)N + n]);
  }
}

// plain fp32 -> bf16 elementwise (for obs_rep)
__global__ void aconv_kernel(const float* __restrict__ src,
                             unsigned short* __restrict__ dst, size_t n) {
  for (size_t i = (size_t)blockIdx.x * blockDim.x + threadIdx.x; i < n;
       i += (size_t)gridDim.x * blockDim.x)
    dst[i] = f2bf(src[i]);
}

// ---------------------------------------------------------------------------
// bf16 WMMA GEMM:  C[M,N] = A[M,512](bf16) @ W[512,N] + bias, W as bf16 Wt[N,512].
// Block = 256 thr = 8 waves; block tile 128x64; wave owns a 16x64 strip.
// Wt tile double-buffered through LDS (one barrier / k-step); K fully unrolled.
// ---------------------------------------------------------------------------
template <bool GELU, bool OUTF, bool OUTB>
__global__ __launch_bounds__(256)
void gemm_bf16_kernel(const unsigned short* __restrict__ A,
                      const unsigned short* __restrict__ Wt,
                      const float* __restrict__ bias,
                      float* __restrict__ C,
                      unsigned short* __restrict__ Cb,
                      int M, int N) {
  constexpr int K = kD;                         // 512, 16 k-steps
  __shared__ alignas(16) unsigned short sB[2][64 * 32];

  const int tid  = threadIdx.x;
  const int wave = tid >> 5;
  const int lane = tid & 31;
  const int l16  = lane & 15;
  const int lhi  = lane >> 4;
  const int m0   = blockIdx.x * 128 + wave * 16;
  const int n0   = blockIdx.y * 64;

  v8f acc[4];
#pragma unroll
  for (int j = 0; j < 4; ++j) acc[j] = vzero8();

  const unsigned short* Arow = A + (size_t)(m0 + l16) * K;

  // cooperative B-stage mapping: thread -> (row 0..63, 16B chunk 0..3)
  const int brow = tid >> 2;
  const int bchk = (tid & 3) * 8;
  const unsigned short* Wrow = Wt + (size_t)(n0 + brow) * K + bchk;

  ushort8 stage = *(const ushort8*)(Wrow);      // prefetch k-step 0

#pragma unroll
  for (int it = 0; it < K / 32; ++it) {
    const int buf = it & 1;
    const int k0  = it * 32;
    *(ushort8*)&sB[buf][brow * 32 + bchk] = stage;
    __syncthreads();
    if (it + 1 < K / 32)
      stage = *(const ushort8*)(Wrow + (it + 1) * 32);   // overlap next fetch

    // A fragment (16x32 bf16, contiguous K-ordering: lanes 0-15 K=[0..15])
    BF16Frag af;
    af.h[0] = *(const ushort8*)(Arow + k0 + lhi * 16);
    af.h[1] = *(const ushort8*)(Arow + k0 + 8 + lhi * 16);

    BF16Frag bf[4];
#pragma unroll
    for (int j = 0; j < 4; ++j) {
      const unsigned short* p = &sB[buf][(j * 16 + l16) * 32 + lhi * 16];
      bf[j].h[0] = *(const ushort8*)(p);
      bf[j].h[1] = *(const ushort8*)(p + 8);
    }
#pragma unroll
    for (int j = 0; j < 4; ++j)
      acc[j] = __builtin_amdgcn_wmma_f32_16x16x32_bf16(
          false, af.v, false, bf[j].v, (short)0, acc[j], false, false);
  }

#pragma unroll
  for (int j = 0; j < 4; ++j) {
    const int   col = n0 + j * 16 + l16;
    const float bb  = bias[col];
#pragma unroll
    for (int r = 0; r < 8; ++r) {
      float v = acc[j][r] + bb;
      if constexpr (GELU) v = gelu_exact(v);
      const int row = m0 + r + lhi * 8;
      if constexpr (OUTF) C[(size_t)row * N + col]  = v;
      if constexpr (OUTB) Cb[(size_t)row * N + col] = f2bf(v);
    }
  }
}

// ---------------------------------------------------------------------------
// Fused causal attention per (batch, head), all-bf16 operands.
// Block = 128 thr = 4 waves; wave owns a 16-row strip of the 64x64 tile.
// ---------------------------------------------------------------------------
__global__ __launch_bounds__(128)
void mha_attn_kernel(const unsigned short* __restrict__ Qg,
                     const unsigned short* __restrict__ Kg,
                     const unsigned short* __restrict__ Vg,
                     unsigned short* __restrict__ Yg) {
  __shared__ alignas(16) unsigned short sQ [64 * 64];
  __shared__ alignas(16) unsigned short sK [64 * 64];
  __shared__ alignas(16) unsigned short sVt[64 * 64];
  __shared__ alignas(16) unsigned short sP [64 * 64];
  __shared__ alignas(16) float          sS [64 * 64];

  const int bh  = blockIdx.x;
  const int b   = bh >> 3;
  const int h   = bh & 7;
  const int tid = threadIdx.x;
  const size_t base = (size_t)(b * kN) * kD + (size_t)h * kHD;

  // vectorized staging: 4096 bf16 per matrix, 4 x ushort8 per thread
  for (int i = tid * 8; i < 64 * 64; i += 128 * 8) {
    const int r = i >> 6, c = i & 63;
    const size_t g = base + (size_t)r * kD + c;
    *(ushort8*)&sQ[i] = *(const ushort8*)(Qg + g);
    *(ushort8*)&sK[i] = *(const ushort8*)(Kg + g);
    ushort8 vv = *(const ushort8*)(Vg + g);
#pragma unroll
    for (int e = 0; e < 8; ++e) sVt[(c + e) * 64 + r] = vv[e];  // transpose
  }
  __syncthreads();

  const int wave = tid >> 5;
  const int lane = tid & 31;
  const int l16  = lane & 15;
  const int lhi  = lane >> 4;
  const int mrow = wave * 16;

  // ---- S = (Q K^T) / sqrt(hd), causal ----
  v8f acc[4];
#pragma unroll
  for (int j = 0; j < 4; ++j) acc[j] = vzero8();

#pragma unroll
  for (int it = 0; it < 2; ++it) {
    const int k0 = it * 32;
    BF16Frag af;
    const unsigned short* qa = &sQ[(mrow + l16) * 64 + k0 + lhi * 16];
    af.h[0] = *(const ushort8*)(qa);
    af.h[1] = *(const ushort8*)(qa + 8);
    BF16Frag bf[4];
#pragma unroll
    for (int j = 0; j < 4; ++j) {
      const unsigned short* kb = &sK[(j * 16 + l16) * 64 + k0 + lhi * 16];
      bf[j].h[0] = *(const ushort8*)(kb);
      bf[j].h[1] = *(const ushort8*)(kb + 8);
    }
#pragma unroll
    for (int j = 0; j < 4; ++j)
      acc[j] = __builtin_amdgcn_wmma_f32_16x16x32_bf16(
          false, af.v, false, bf[j].v, (short)0, acc[j], false, false);
  }
#pragma unroll
  for (int j = 0; j < 4; ++j) {
    const int col = j * 16 + l16;
#pragma unroll
    for (int r = 0; r < 8; ++r) {
      const int row = mrow + r + lhi * 8;
      const float s = acc[j][r] * 0.125f;                 // 1/sqrt(64)
      sS[row * 64 + col] = (col <= row) ? s : -3.0e38f;   // causal mask
    }
  }
  __syncthreads();

  // ---- row softmax (fp32) -> bf16 probs ----
  if (tid < 64) {
    float mx = -3.4e38f;
    for (int c = 0; c < 64; ++c) mx = fmaxf(mx, sS[tid * 64 + c]);
    float sum = 0.0f;
    for (int c = 0; c < 64; ++c) {
      const float e = __expf(sS[tid * 64 + c] - mx);
      sS[tid * 64 + c] = e;
      sum += e;
    }
    const float inv = 1.0f / sum;
    for (int c = 0; c < 64; ++c) sP[tid * 64 + c] = f2bf(sS[tid * 64 + c] * inv);
  }
  __syncthreads();

  // ---- O = P @ V ----
  v8f oc[4];
#pragma unroll
  for (int j = 0; j < 4; ++j) oc[j] = vzero8();

#pragma unroll
  for (int it = 0; it < 2; ++it) {
    const int k0 = it * 32;
    BF16Frag af;
    const unsigned short* pa = &sP[(mrow + l16) * 64 + k0 + lhi * 16];
    af.h[0] = *(const ushort8*)(pa);
    af.h[1] = *(const ushort8*)(pa + 8);
    BF16Frag bf[4];
#pragma unroll
    for (int j = 0; j < 4; ++j) {
      const unsigned short* vb = &sVt[(j * 16 + l16) * 64 + k0 + lhi * 16];
      bf[j].h[0] = *(const ushort8*)(vb);
      bf[j].h[1] = *(const ushort8*)(vb + 8);
    }
#pragma unroll
    for (int j = 0; j < 4; ++j)
      oc[j] = __builtin_amdgcn_wmma_f32_16x16x32_bf16(
          false, af.v, false, bf[j].v, (short)0, oc[j], false, false);
  }
#pragma unroll
  for (int j = 0; j < 4; ++j) {
    const int col = j * 16 + l16;
#pragma unroll
    for (int r = 0; r < 8; ++r) {
      const int row = mrow + r + lhi * 8;
      Yg[base + (size_t)row * kD + col] = f2bf(oc[j][r]);
    }
  }
}

// ---------------------------------------------------------------------------
// Out = LayerNorm(X + R)  (R nullable); writes fp32 Out and bf16 Outb.
// ---------------------------------------------------------------------------
__global__ __launch_bounds__(128)
void resid_ln_kernel(const float* __restrict__ X, const float* __restrict__ R,
                     const float* __restrict__ g, const float* __restrict__ be,
                     float* __restrict__ Out, unsigned short* __restrict__ Outb) {
  __shared__ float red[4];
  const int row = blockIdx.x;
  const int tid = threadIdx.x;
  const float* xr = X + (size_t)row * kD;
  const float* rr = R ? R + (size_t)row * kD : nullptr;

  float v[4];
  float s = 0.0f;
#pragma unroll
  for (int i = 0; i < 4; ++i) {
    const int c = tid + i * 128;
    float t = xr[c];
    if (rr) t += rr[c];
    v[i] = t;
    s += t;
  }
  const float mean = block128_sum(s, red) * (1.0f / kD);
  float s2 = 0.0f;
#pragma unroll
  for (int i = 0; i < 4; ++i) {
    const float d = v[i] - mean;
    s2 += d * d;
  }
  const float var  = block128_sum(s2, red) * (1.0f / kD);
  const float rstd = rsqrtf(var + 1e-5f);
#pragma unroll
  for (int i = 0; i < 4; ++i) {
    const int c = tid + i * 128;
    const float o = (v[i] - mean) * rstd * g[c] + be[c];
    Out [(size_t)row * kD + c] = o;
    Outb[(size_t)row * kD + c] = f2bf(o);
  }
}

// ---------------------------------------------------------------------------
// Encoder: X = LN(gelu(action @ w_ae)); writes fp32 X and bf16 Xb.
// ---------------------------------------------------------------------------
__global__ __launch_bounds__(128)
void encoder_kernel(const float* __restrict__ act, const float* __restrict__ Wae,
                    const float* __restrict__ g, const float* __restrict__ be,
                    float* __restrict__ X, unsigned short* __restrict__ Xb) {
  __shared__ float sa[65];
  __shared__ float red[4];
  const int row = blockIdx.x;
  const int tid = threadIdx.x;
  if (tid < 65) sa[tid] = act[(size_t)row * 65 + tid];
  __syncthreads();

  float o[4];
  float s = 0.0f;
#pragma unroll
  for (int i = 0; i < 4; ++i) {
    const int n = tid + i * 128;
    float a = 0.0f;
    for (int k = 0; k < 65; ++k) a += sa[k] * Wae[k * kD + n];
    o[i] = gelu_exact(a);
    s += o[i];
  }
  const float mean = block128_sum(s, red) * (1.0f / kD);
  float s2 = 0.0f;
#pragma unroll
  for (int i = 0; i < 4; ++i) {
    const float d = o[i] - mean;
    s2 += d * d;
  }
  const float var  = block128_sum(s2, red) * (1.0f / kD);
  const float rstd = rsqrtf(var + 1e-5f);
#pragma unroll
  for (int i = 0; i < 4; ++i) {
    const int n = tid + i * 128;
    const float y = (o[i] - mean) * rstd * g[n] + be[n];
    X [(size_t)row * kD + n] = y;
    Xb[(size_t)row * kD + n] = f2bf(y);
  }
}

// ---------------------------------------------------------------------------
template <bool GELU, bool OUTF, bool OUTB>
static void launch_gemm(const unsigned short* A, const unsigned short* Wt,
                        const float* bias, float* C, unsigned short* Cb,
                        int M, int N, hipStream_t s) {
  dim3 grid(M / 128, N / 64), blk(256);
  gemm_bf16_kernel<GELU, OUTF, OUTB><<<grid, blk, 0, s>>>(A, Wt, bias, C, Cb, M, N);
}

extern "C" void kernel_launch(void* const* d_in, const int* in_sizes, int n_in,
                              void* d_out, int out_size, void* d_ws, size_t ws_size,
                              hipStream_t stream) {
  const float* action  = (const float*)d_in[0];
  const float* obs_rep = (const float*)d_in[1];
  const float* w_ae    = (const float*)d_in[3];
  const float* ln0_g   = (const float*)d_in[4];
  const float* ln0_b   = (const float*)d_in[5];
  const float* bln_g   = (const float*)d_in[6];
  const float* bln_b   = (const float*)d_in[7];
  const float* bwq     = (const float*)d_in[8];
  const float* bwk     = (const float*)d_in[9];
  const float* bwv     = (const float*)d_in[10];
  const float* bwp     = (const float*)d_in[11];
  const float* bbq     = (const float*)d_in[12];
  const float* bbk     = (const float*)d_in[13];
  const float* bbv     = (const float*)d_in[14];
  const float* bbp     = (const float*)d_in[15];
  const float* bw1     = (const float*)d_in[16];
  const float* bb1     = (const float*)d_in[17];
  const float* bw2     = (const float*)d_in[18];
  const float* bb2     = (const float*)d_in[19];
  const float* wh1     = (const float*)d_in[20];
  const float* bh1     = (const float*)d_in[21];
  const float* lnh_g   = (const float*)d_in[22];
  const float* lnh_b   = (const float*)d_in[23];
  const float* wh2     = (const float*)d_in[24];
  const float* bh2     = (const float*)d_in[25];
  float* out = (float*)d_out;

  // ---- workspace carve-up ----
  char* ws = (char*)d_ws;
  const size_t actF = (size_t)kBN * kD * sizeof(float);
  const size_t actB = (size_t)kBN * kD * sizeof(unsigned short);
  float* X  = (float*)ws; ws += actF;           // persistent activation fp32
  float* T  = (float*)ws; ws += actF;           // proj / mlp2 / head1 fp32
  unsigned short* Xb  = (unsigned short*)ws; ws += actB;
  unsigned short* ORb = (unsigned short*)ws; ws += actB;  // obs_rep bf16
  unsigned short* Qb  = (unsigned short*)ws; ws += actB;
  unsigned short* Kb  = (unsigned short*)ws; ws += actB;
  unsigned short* Vb  = (unsigned short*)ws; ws += actB;
  unsigned short* Yb  = (unsigned short*)ws; ws += actB;  // attention out bf16
  unsigned short* wbf = (unsigned short*)ws;

  const size_t MAT = (size_t)kD * kD;
  unsigned short* wq_t  = wbf;
  unsigned short* wk_t  = wq_t  + 8 * MAT;
  unsigned short* wv_t  = wk_t  + 8 * MAT;
  unsigned short* wp_t  = wv_t  + 8 * MAT;
  unsigned short* w1_t  = wp_t  + 8 * MAT;
  unsigned short* w2_t  = w1_t  + 4 * MAT;
  unsigned short* wh1_t = w2_t  + 4 * MAT;
  unsigned short* wh2_t = wh1_t + MAT;

  // ---- weight / input conversion ----
  wconv_kernel<<<2048, 256, 0, stream>>>(bwq, wq_t, kD, kD, 8);
  wconv_kernel<<<2048, 256, 0, stream>>>(bwk, wk_t, kD, kD, 8);
  wconv_kernel<<<2048, 256, 0, stream>>>(bwv, wv_t, kD, kD, 8);
  wconv_kernel<<<2048, 256, 0, stream>>>(bwp, wp_t, kD, kD, 8);
  wconv_kernel<<<1024, 256, 0, stream>>>(bw1, w1_t, kD, kD, 4);
  wconv_kernel<<<1024, 256, 0, stream>>>(bw2, w2_t, kD, kD, 4);
  wconv_kernel<<<512,  256, 0, stream>>>(wh1, wh1_t, kD, kD, 1);
  wconv_kernel<<<128,  256, 0, stream>>>(wh2, wh2_t, kD, kA, 1);
  aconv_kernel<<<2048, 256, 0, stream>>>(obs_rep, ORb, (size_t)kBN * kD);

  // ---- action encoder ----
  encoder_kernel<<<kBN, 128, 0, stream>>>(action, w_ae, ln0_g, ln0_b, X, Xb);

  for (int i = 0; i < kNB; ++i) {
    const float* g0 = bln_g + (size_t)(i * 3 + 0) * kD;
    const float* e0 = bln_b + (size_t)(i * 3 + 0) * kD;
    const float* g1 = bln_g + (size_t)(i * 3 + 1) * kD;
    const float* e1 = bln_b + (size_t)(i * 3 + 1) * kD;
    const float* g2 = bln_g + (size_t)(i * 3 + 2) * kD;
    const float* e2 = bln_b + (size_t)(i * 3 + 2) * kD;

    // sublayer 1: causal self-attention on x
    {
      const int p = i * 2 + 0;
      launch_gemm<false, false, true>(Xb, wq_t + p * MAT, bbq + p * kD, nullptr, Qb, kBN, kD, stream);
      launch_gemm<false, false, true>(Xb, wk_t + p * MAT, bbk + p * kD, nullptr, Kb, kBN, kD, stream);
      launch_gemm<false, false, true>(Xb, wv_t + p * MAT, bbv + p * kD, nullptr, Vb, kBN, kD, stream);
      mha_attn_kernel<<<kB * kH, 128, 0, stream>>>(Qb, Kb, Vb, Yb);
      launch_gemm<false, true, false>(Yb, wp_t + p * MAT, bbp + p * kD, T, nullptr, kBN, kD, stream);
      resid_ln_kernel<<<kBN, 128, 0, stream>>>(X, T, g0, e0, X, Xb);
    }
    // sublayer 2: causal cross-attention (q = obs_rep, k/v = x)
    {
      const int p = i * 2 + 1;
      launch_gemm<false, false, true>(ORb, wq_t + p * MAT, bbq + p * kD, nullptr, Qb, kBN, kD, stream);
      launch_gemm<false, false, true>(Xb,  wk_t + p * MAT, bbk + p * kD, nullptr, Kb, kBN, kD, stream);
      launch_gemm<false, false, true>(Xb,  wv_t + p * MAT, bbv + p * kD, nullptr, Vb, kBN, kD, stream);
      mha_attn_kernel<<<kB * kH, 128, 0, stream>>>(Qb, Kb, Vb, Yb);
      launch_gemm<false, true, false>(Yb, wp_t + p * MAT, bbp + p * kD, T, nullptr, kBN, kD, stream);
      resid_ln_kernel<<<kBN, 128, 0, stream>>>(obs_rep, T, g1, e1, X, Xb);
    }
    // sublayer 3: GELU MLP
    {
      launch_gemm<true, false, true>(Xb, w1_t + i * MAT, bb1 + (size_t)i * kD, nullptr, Qb, kBN, kD, stream);
      launch_gemm<false, true, false>(Qb, w2_t + i * MAT, bb2 + (size_t)i * kD, T, nullptr, kBN, kD, stream);
      resid_ln_kernel<<<kBN, 128, 0, stream>>>(X, T, g2, e2, X, Xb);
    }
  }

  // ---- head: LN(gelu(x@wh1+bh1)) @ wh2 + bh2 ----
  launch_gemm<true, true, false>(Xb, wh1_t, bh1, T, nullptr, kBN, kD, stream);
  resid_ln_kernel<<<kBN, 128, 0, stream>>>(T, (const float*)nullptr, lnh_g, lnh_b, T, Qb);
  launch_gemm<false, true, false>(Qb, wh2_t, bh2, out, nullptr, kBN, kA, stream);
}
